// ScaledDotProductAttention_36764920054096
// MI455X (gfx1250) — compile-verified, matrix operand order checked by reading, not run on hardware
//
#include <hip/hip_runtime.h>
#include <hip/hip_bf16.h>

// ---- problem constants (from reference) ----
#define B_DIM 2
#define H_DIM 16
#define S_DIM 2048
#define D_DIM 64
// log2(e)/sqrt(64) folded into Q's bf16 scale: QK^T scores arrive in base-2
// units so every softmax exponential is a single native v_exp_f32.
#define Q_SCALE (0.125f * 1.44269504088896340736f)

typedef __attribute__((ext_vector_type(16))) __bf16 v16bf;
typedef __attribute__((ext_vector_type(8)))  float  v8f;

// raw v_exp_f32 / v_log_f32 (base 2), no OCML denormal guards
#define EXP2(x) __builtin_amdgcn_exp2f(x)
#define LOG2(x) __builtin_amdgcn_logf(x)

// Load a 16-bit A/B fragment from LDS following the ISA §7.12.2 striping:
// element e = g*8+i  <->  K-offset g*16 + khalf*8 + i   (khalf = lane>>4)
__device__ __forceinline__ v16bf frag_from_lds(const __bf16* p, int khalf) {
  v16bf r;
#pragma unroll
  for (int g = 0; g < 2; ++g)
#pragma unroll
    for (int i = 0; i < 8; ++i)
      r[g * 8 + i] = p[g * 16 + khalf * 8 + i];
  return r;
}

__device__ __forceinline__ void cvt8(__bf16* d, float4 x, float4 y) {
  d[0]=(__bf16)x.x; d[1]=(__bf16)x.y; d[2]=(__bf16)x.z; d[3]=(__bf16)x.w;
  d[4]=(__bf16)y.x; d[5]=(__bf16)y.y; d[6]=(__bf16)y.z; d[7]=(__bf16)y.w;
}

__global__ __launch_bounds__(256)
void sdpa_fwd_kernel(const float* __restrict__ qg,
                     const float* __restrict__ kg,
                     const float* __restrict__ vg,
                     const int*   __restrict__ maskg,
                     float* __restrict__ outp,
                     float* __restrict__ attnp) {
  // Double-buffered LDS staging: K 2x(32x64) bf16, V^T 2x(64x32) bf16,
  // per-wave P tiles 16x32 bf16. 24KB of the 320KB WGP pool.
  __shared__ __bf16 ktile[2][32 * 64];
  __shared__ __bf16 vtile[2][64 * 32];
  __shared__ __bf16 pbuf[8][16 * 32];

  const int tid    = threadIdx.x;
  const int wave   = tid >> 5;
  const int lane   = tid & 31;
  const int lane16 = lane & 15;
  const int khalf  = lane >> 4;

  const int bh     = blockIdx.x >> 4;            // 0..31  (b*H+h)
  const int qchunk = blockIdx.x & 15;            // 16 chunks of 128 q rows
  const int qrow0  = qchunk * 128 + wave * 16;   // this wave's 16 q rows
  const int b      = bh >> 4;

  const float* qbase = qg + ((size_t)bh * S_DIM) * D_DIM;
  const float* kbase = kg + ((size_t)bh * S_DIM) * D_DIM;
  const float* vbase = vg + ((size_t)bh * S_DIM) * D_DIM;
  const int*   maskp = maskg + (size_t)b * S_DIM;
  float* attn_bh = attnp + (size_t)bh * S_DIM * S_DIM;

  // ---- load Q tile (16x64) into two bf16 A fragments, pre-scaled ----
  const float* qrow = qbase + (size_t)(qrow0 + lane16) * D_DIM;
  v16bf aq[2];
#pragma unroll
  for (int f = 0; f < 2; ++f)
#pragma unroll
    for (int g = 0; g < 2; ++g) {
      const float4 x = *(const float4*)(qrow + 32 * f + g * 16 + khalf * 8);
      const float4 y = *(const float4*)(qrow + 32 * f + g * 16 + khalf * 8 + 4);
      aq[f][g * 8 + 0] = (__bf16)(x.x * Q_SCALE);
      aq[f][g * 8 + 1] = (__bf16)(x.y * Q_SCALE);
      aq[f][g * 8 + 2] = (__bf16)(x.z * Q_SCALE);
      aq[f][g * 8 + 3] = (__bf16)(x.w * Q_SCALE);
      aq[f][g * 8 + 4] = (__bf16)(y.x * Q_SCALE);
      aq[f][g * 8 + 5] = (__bf16)(y.y * Q_SCALE);
      aq[f][g * 8 + 6] = (__bf16)(y.z * Q_SCALE);
      aq[f][g * 8 + 7] = (__bf16)(y.w * Q_SCALE);
    }

  const int krow = tid >> 3;   // 0..31 (key within chunk)
  const int kgrp = tid & 7;    // 0..7  (group of 8 floats in D)
  const int NCH  = S_DIM / 32;

  // Per-lane online-softmax state; lane covers keys == lane16 (mod 16).
  // C-layout: VGPR r holds row M = r + 8*khalf.
  // pm tracks the max of RAW scores (upper bound of masked max -- softmax is
  // invariant to the shift as long as l uses the same shift), so no cndmask
  // is needed on the max path; the mask enters as a {0,1} float multiply.
  float pm[8], pl[8];
#pragma unroll
  for (int r = 0; r < 8; ++r) { pm[r] = -3.0e38f; pl[r] = 0.0f; }

  // ======= pass 1: per-lane online max/sum (base-2, shuffle-free) =========
  {
    {
      const float* kr = kbase + (size_t)krow * D_DIM + kgrp * 8;
      cvt8(&ktile[0][krow * 64 + kgrp * 8], *(const float4*)kr, *(const float4*)(kr + 4));
    }
    __syncthreads();
    int cur = 0;
    for (int kc = 0; kc < NCH; ++kc, cur ^= 1) {
      // issue global loads for next chunk (wait sinks past the WMMA block)
      float4 nkx, nky;
      const bool has_next = (kc + 1 < NCH);
      if (has_next) {
        const float* kr = kbase + (size_t)((kc + 1) * 32 + krow) * D_DIM + kgrp * 8;
        nkx = *(const float4*)kr;
        nky = *(const float4*)(kr + 4);
      }
#pragma unroll
      for (int t = 0; t < 2; ++t) {
        const int kb  = kc * 32 + t * 16;
        const int key = t * 16 + lane16;
        v16bf b0 = frag_from_lds(&ktile[cur][key * 64 +  0], khalf);
        v16bf b1 = frag_from_lds(&ktile[cur][key * 64 + 32], khalf);
        v8f c = {};
        c = __builtin_amdgcn_wmma_f32_16x16x32_bf16(false, aq[0], false, b0, (short)0, c, false, false);
        c = __builtin_amdgcn_wmma_f32_16x16x32_bf16(false, aq[1], false, b1, (short)0, c, false, false);
        const float mkf = maskp[kb + lane16] ? 1.0f : 0.0f;
#pragma unroll
        for (int r = 0; r < 8; ++r) {
          float s  = c[r];
          float nm = fmaxf(pm[r], s);
          pl[r] = pl[r] * EXP2(pm[r] - nm) + mkf * EXP2(s - nm);
          pm[r] = nm;
        }
      }
      if (has_next)
        cvt8(&ktile[cur ^ 1][krow * 64 + kgrp * 8], nkx, nky);
      __syncthreads();
    }
  }

  // single cross-lane (16-lane) merge of (m, l) pairs per row, then fold the
  // normalizer into the shift:  p = 2^(s - (m + log2 l)) -> one v_exp per prob
  float mb[8];
#pragma unroll
  for (int r = 0; r < 8; ++r) {
    float ml = pm[r], ll = pl[r];
#pragma unroll
    for (int o = 8; o > 0; o >>= 1) {
      float om = __shfl_xor(ml, o, 16);
      float ol = __shfl_xor(ll, o, 16);
      float nm = fmaxf(ml, om);
      ll = ll * EXP2(ml - nm) + ol * EXP2(om - nm);
      ml = nm;
    }
    mb[r] = ml + LOG2(ll);
  }

  // out accumulators: 4 d-tiles of 16x16 f32
  v8f acc[4] = {};
  __bf16* pw = &pbuf[wave][0];

  // ============ pass 2: recompute scores, emit attn, P·V ============
  {
    {
      const float* kr = kbase + (size_t)krow * D_DIM + kgrp * 8;
      cvt8(&ktile[0][krow * 64 + kgrp * 8], *(const float4*)kr, *(const float4*)(kr + 4));
      const float* vr = vbase + (size_t)krow * D_DIM + kgrp * 8;
      const float4 x = *(const float4*)vr;
      const float4 y = *(const float4*)(vr + 4);
      __bf16* vt = &vtile[0][(kgrp * 8) * 32 + krow];   // transposed: [d][key]
      vt[0*32]=(__bf16)x.x; vt[1*32]=(__bf16)x.y; vt[2*32]=(__bf16)x.z; vt[3*32]=(__bf16)x.w;
      vt[4*32]=(__bf16)y.x; vt[5*32]=(__bf16)y.y; vt[6*32]=(__bf16)y.z; vt[7*32]=(__bf16)y.w;
    }
    __syncthreads();
    int cur = 0;
    for (int kc = 0; kc < NCH; ++kc, cur ^= 1) {
      float4 nkx, nky, nvx, nvy;
      const bool has_next = (kc + 1 < NCH);
      if (has_next) {
        const float* kr = kbase + (size_t)((kc + 1) * 32 + krow) * D_DIM + kgrp * 8;
        nkx = *(const float4*)kr;
        nky = *(const float4*)(kr + 4);
        const float* vr = vbase + (size_t)((kc + 1) * 32 + krow) * D_DIM + kgrp * 8;
        nvx = *(const float4*)vr;
        nvy = *(const float4*)(vr + 4);
      }
#pragma unroll
      for (int t = 0; t < 2; ++t) {
        const int kb  = kc * 32 + t * 16;
        const int key = t * 16 + lane16;
        v16bf b0 = frag_from_lds(&ktile[cur][key * 64 +  0], khalf);
        v16bf b1 = frag_from_lds(&ktile[cur][key * 64 + 32], khalf);
        v8f c = {};
        c = __builtin_amdgcn_wmma_f32_16x16x32_bf16(false, aq[0], false, b0, (short)0, c, false, false);
        c = __builtin_amdgcn_wmma_f32_16x16x32_bf16(false, aq[1], false, b1, (short)0, c, false, false);
        const float mkf = maskp[kb + lane16] ? 1.0f : 0.0f;
#pragma unroll
        for (int r = 0; r < 8; ++r) {
          float p = mkf * EXP2(c[r] - mb[r]);          // normalized prob
          const int Ml = r + 8 * khalf;
          // streaming NT store: attn is write-once, keep L2 for K/V reuse
          __builtin_nontemporal_store(p, &attn_bh[(size_t)(qrow0 + Ml) * S_DIM + kb + lane16]);
          pw[Ml * 32 + t * 16 + lane16] = (__bf16)p;   // stage for PV matmul
        }
      }

      // A fragment of P (16 x 32 keys) from per-wave LDS region
      v16bf pa = frag_from_lds(&pw[lane16 * 32], khalf);
#pragma unroll
      for (int dt = 0; dt < 4; ++dt) {
        v16bf bv = frag_from_lds(&vtile[cur][(dt * 16 + lane16) * 32], khalf);
        acc[dt] = __builtin_amdgcn_wmma_f32_16x16x32_bf16(false, pa, false, bv, (short)0, acc[dt], false, false);
      }

      if (has_next) {
        cvt8(&ktile[cur ^ 1][krow * 64 + kgrp * 8], nkx, nky);
        __bf16* vt = &vtile[cur ^ 1][(kgrp * 8) * 32 + krow];
        vt[0*32]=(__bf16)nvx.x; vt[1*32]=(__bf16)nvx.y; vt[2*32]=(__bf16)nvx.z; vt[3*32]=(__bf16)nvx.w;
        vt[4*32]=(__bf16)nvy.x; vt[5*32]=(__bf16)nvy.y; vt[6*32]=(__bf16)nvy.z; vt[7*32]=(__bf16)nvy.w;
      }
      __syncthreads();
    }
  }

  // ---- write out tile (16 x 64), NT (write-once) ----
#pragma unroll
  for (int dt = 0; dt < 4; ++dt)
#pragma unroll
    for (int r = 0; r < 8; ++r)
      __builtin_nontemporal_store(
          acc[dt][r],
          &outp[((size_t)bh * S_DIM + qrow0 + r + 8 * khalf) * D_DIM + dt * 16 + lane16]);
}

extern "C" void kernel_launch(void* const* d_in, const int* in_sizes, int n_in,
                              void* d_out, int out_size, void* d_ws, size_t ws_size,
                              hipStream_t stream) {
  (void)in_sizes; (void)n_in; (void)out_size; (void)d_ws; (void)ws_size;
  const float* q = (const float*)d_in[0];
  const float* k = (const float*)d_in[1];
  const float* v = (const float*)d_in[2];
  const int* mask = (const int*)d_in[3];
  float* outp  = (float*)d_out;                                   // [B,H,S,D]
  float* attnp = outp + (size_t)B_DIM * H_DIM * S_DIM * D_DIM;    // [B,H,S,S]

  const int blocks = B_DIM * H_DIM * (S_DIM / 128);               // 512
  sdpa_fwd_kernel<<<blocks, 256, 0, stream>>>(q, k, v, mask, outp, attnp);
}
